// GGUFLinear_53077205844002
// MI455X (gfx1250) — matrix operand order repaired and checked by reading, not compiled, not run on hardware
//
#include <hip/hip_runtime.h>
#include <stdint.h>

#define IN_F      4096
#define OUT_F     14336
#define QK_K      256
#define BLK_BYTES 210
#define N_CHUNKS  (IN_F / QK_K)   // 16

#define WAVES_PER_WG 4
#define THREADS      (WAVES_PER_WG * 32)
#define SPLIT_K      4
#define CH_PER_SPLIT (N_CHUNKS / SPLIT_K)   // 4

typedef __attribute__((ext_vector_type(16))) _Float16 v16h;
typedef __attribute__((ext_vector_type(8)))  float    v8f;
typedef __attribute__((ext_vector_type(2)))  _Float16 h2;

__device__ __forceinline__ uint32_t dup16(_Float16 h) {
    uint32_t u = __builtin_bit_cast(unsigned short, h);
    return u | (u << 16);
}

__global__ __launch_bounds__(THREADS)
void q6k_gemm_wmma_f16(const float* __restrict__ x,
                       const uint8_t* __restrict__ wq,
                       const float* __restrict__ bias,
                       float* __restrict__ dst,      // out (addBias=1) or split-K partials
                       int nChunks, int addBias)
{
    __shared__ _Float16 sA[16][QK_K];                 // 8 KB  activations (f16)
    __shared__ _Float16 sB[WAVES_PER_WG][16][QK_K];   // 32 KB dequantized weights

    const int tid  = threadIdx.x;
    const int lane = tid & 31;
    const int w    = tid >> 5;

    const int nBase  = (blockIdx.x * WAVES_PER_WG + w) * 16;  // N tile
    const int cBegin = blockIdx.y * nChunks;                  // K-split start

    const int row  = lane >> 1;   // dequant: 2 lanes per N row
    const int half = lane & 1;    // which 128-elt half of the Q6_K block

    const int xm = tid >> 3;          // x staging: token
    const int xc = (tid & 7) * 32;    // x staging: column base

    v8f acc = {};

    for (int c = 0; c < nChunks; ++c) {
        const int chunk = cBegin + c;
        const int k0 = chunk * QK_K;

        // ---- stage activations: 16x256 f32 -> f16 via v_cvt_pkrtz ----
        {
            const float4* xp4 = (const float4*)(x + (size_t)xm * IN_F + k0 + xc);
            uint32_t* dstA = (uint32_t*)&sA[xm][xc];
            #pragma unroll
            for (int v = 0; v < 8; ++v) {
                float4 f = xp4[v];
                dstA[2 * v]     = __builtin_bit_cast(uint32_t, __builtin_amdgcn_cvt_pkrtz(f.x, f.y));
                dstA[2 * v + 1] = __builtin_bit_cast(uint32_t, __builtin_amdgcn_cvt_pkrtz(f.z, f.w));
            }
        }

        // ---- packed-SIMD Q6_K dequant into sB ----
        {
            const size_t blk = (size_t)(nBase + row) * N_CHUNKS + chunk;
            const uint8_t* bp = wq + blk * BLK_BYTES;

            uint32_t qlw[16]; __builtin_memcpy(qlw, bp + half * 64, 64);
            uint32_t qhw[8];  __builtin_memcpy(qhw, bp + 128 + half * 32, 32);
            uint32_t scw[2];  __builtin_memcpy(scw, bp + 192 + half * 8, 8);
            uint16_t du;      __builtin_memcpy(&du, bp + 208, 2);
            const float d = (float)__builtin_bit_cast(_Float16, du);

            if (c + 1 < nChunks) __builtin_prefetch(bp + BLK_BYTES, 0, 0);

            // per-16-group scale pairs: r = (1024+v)*s + (-1056*s) == (v-32)*s
            h2 sc2[8], of2[8];
            #pragma unroll
            for (int i = 0; i < 8; ++i) {
                int8_t s8 = (int8_t)((scw[i >> 2] >> ((i & 3) * 8)) & 0xff);
                float s = d * (float)s8;
                _Float16 sh = (_Float16)s;
                _Float16 oh = (_Float16)(s * -1056.0f);
                sc2[i] = (h2){sh, sh};
                of2[i] = (h2){oh, oh};
            }

            _Float16* dstRow = &sB[w][row][half * 128];

            #pragma unroll
            for (int i = 0; i < 8; ++i) {           // 4 bytes (b=4i..4i+3) per iter
                const uint32_t L0 = qlw[i];         // ql[4i..]
                const uint32_t L1 = qlw[8 + i];     // ql[32+4i..]
                const uint32_t H  = qhw[i];         // qh[4i..]

                uint32_t qb[4];
                qb[0] = (L0 & 0x0f0f0f0fu)        | ((H << 4) & 0x30303030u);
                qb[1] = (L1 & 0x0f0f0f0fu)        | ((H << 2) & 0x30303030u);
                qb[2] = ((L0 >> 4) & 0x0f0f0f0fu) | (H        & 0x30303030u);
                qb[3] = ((L1 >> 4) & 0x0f0f0f0fu) | ((H >> 2) & 0x30303030u);

                const int si = i >> 2;              // 16-group within 32-run
                #pragma unroll
                for (int g = 0; g < 4; ++g) {
                    const h2 s2 = sc2[g * 2 + si];
                    const h2 o2 = of2[g * 2 + si];
                    // splice bytes against 0x64 -> f16 (1024+v) pairs, one v_perm each
                    uint32_t p01 = __builtin_amdgcn_perm(0x64646464u, qb[g], 0x04010400u);
                    uint32_t p23 = __builtin_amdgcn_perm(0x64646464u, qb[g], 0x04030402u);
                    h2 r01 = __builtin_bit_cast(h2, p01) * s2 + o2;   // v_pk_fma_f16
                    h2 r23 = __builtin_bit_cast(h2, p23) * s2 + o2;
                    uint2 st;
                    st.x = __builtin_bit_cast(uint32_t, r01);
                    st.y = __builtin_bit_cast(uint32_t, r23);
                    *(uint2*)&dstRow[g * 32 + 4 * i] = st;
                }
            }
        }

        __syncthreads();

        // ---- 8 WMMA steps over this 256-wide K chunk ----
        const int sel = lane >> 4;
        const int am  = lane & 15;

        #pragma unroll
        for (int kk = 0; kk < QK_K / 32; ++kk) {
            union { uint4 u[2]; v16h h; } fa, fb;
            const _Float16* ap = &sA[am][kk * 32 + sel * 8];
            fa.u[0] = *(const uint4*)(ap);
            fa.u[1] = *(const uint4*)(ap + 16);
            const _Float16* bp2 = &sB[w][am][kk * 32 + sel * 8];
            fb.u[0] = *(const uint4*)(bp2);
            fb.u[1] = *(const uint4*)(bp2 + 16);

            acc = __builtin_amdgcn_wmma_f32_16x16x32_f16(
                false, fa.h, false, fb.h, (short)0, acc, false, false);
        }

        __syncthreads();
    }

    // ---- epilogue ----
    const int n  = lane & 15;
    const int o  = nBase + n;
    const float bv = addBias ? bias[o] : 0.0f;
    const int mb = (lane >> 4) * 8;
    float* dstBase = dst + (size_t)blockIdx.y * 16 * OUT_F;
    #pragma unroll
    for (int j = 0; j < 8; ++j) {
        dstBase[(size_t)(mb + j) * OUT_F + o] = acc[j] + bv;
    }
}

__global__ __launch_bounds__(256)
void q6k_reduce_split(const float* __restrict__ part,
                      const float* __restrict__ bias,
                      float* __restrict__ out)
{
    const int i = blockIdx.x * 256 + threadIdx.x;   // 0 .. 16*OUT_F
    if (i < 16 * OUT_F) {
        float v = bias[i % OUT_F];
        #pragma unroll
        for (int s = 0; s < SPLIT_K; ++s)
            v += part[(size_t)s * 16 * OUT_F + i];
        out[i] = v;
    }
}

extern "C" void kernel_launch(void* const* d_in, const int* in_sizes, int n_in,
                              void* d_out, int out_size, void* d_ws, size_t ws_size,
                              hipStream_t stream) {
    (void)in_sizes; (void)n_in; (void)out_size;
    const float*   x    = (const float*)d_in[0];
    const uint8_t* wq   = (const uint8_t*)d_in[1];
    const float*   bias = (const float*)d_in[2];
    float*         out  = (float*)d_out;

    const size_t wsNeed = (size_t)SPLIT_K * 16 * OUT_F * sizeof(float);
    const int nTileBlocks = OUT_F / (16 * WAVES_PER_WG);   // 224

    if (ws_size >= wsNeed) {
        // split-K x4: 3584 waves, deterministic partials + reduction
        q6k_gemm_wmma_f16<<<dim3(nTileBlocks, SPLIT_K), THREADS, 0, stream>>>(
            x, wq, bias, (float*)d_ws, CH_PER_SPLIT, 0);
        q6k_reduce_split<<<(16 * OUT_F + 255) / 256, 256, 0, stream>>>(
            (const float*)d_ws, bias, out);
    } else {
        // single-pass fallback
        q6k_gemm_wmma_f16<<<dim3(nTileBlocks, 1), THREADS, 0, stream>>>(
            x, wq, bias, out, N_CHUNKS, 1);
    }
}
